// MeshGraphNet_67937792688581
// MI455X (gfx1250) — matrix operand order repaired and checked by reading, not compiled, not run on hardware
//
#include <hip/hip_runtime.h>

#define NN 40000
#define EE 120000
#define LL 6
#define HH 128
#define DIN 11
#define DEDGE 3
#define DOUT 3

typedef __attribute__((ext_vector_type(16))) __bf16 v16bf;
typedef __attribute__((ext_vector_type(8)))  __bf16 v8bf;
typedef __attribute__((ext_vector_type(8)))  float  v8f;

// ---------- helpers ----------

static __device__ __forceinline__ __bf16 f2bf(float f) {
  unsigned u = __builtin_bit_cast(unsigned, f);
  unsigned r = u + 0x7FFFu + ((u >> 16) & 1u);   // round-to-nearest-even
  unsigned short s = (unsigned short)(r >> 16);
  return __builtin_bit_cast(__bf16, s);
}

static __device__ __forceinline__ v16bf load_afrag(const __bf16* p) {
  v8bf lo = *(const v8bf*)p;
  v8bf hi = *(const v8bf*)(p + 16);
  v16bf a;
#pragma unroll
  for (int j = 0; j < 8; ++j) { a[j] = lo[j]; a[j + 8] = hi[j]; }
  return a;
}

// ---- single row-tile (16 rows) GEMM: used by encoder/decoder ----
// A-frag layout (16-bit A 16x32): elem j -> k = 16*(j>>3) + 8*half + (j&7),
// row = lane&15.  B packed fragment-major: ((ct*KT+kt)*32 + lane)*16 + j.
static __device__ __forceinline__ v8f wmma_rows16(
    const __bf16* ldsA, int strideA, int KT,
    const __bf16* Bp, int ct, float bv) {
  int lane = threadIdx.x & 31;
  int half = lane >> 4;
  int row  = lane & 15;
  v8f c;
#pragma unroll
  for (int r = 0; r < 8; ++r) c[r] = bv;   // bias depends only on col = lane&15
  const __bf16* arow = ldsA + row * strideA + half * 8;
  const __bf16* bp   = Bp + (size_t)ct * KT * 512 + lane * 16;
#pragma unroll 4
  for (int kt = 0; kt < KT; ++kt) {
    v16bf a = load_afrag(arow + kt * 32);
    v16bf b = *(const v16bf*)(bp + kt * 512);
    c = __builtin_amdgcn_wmma_f32_16x16x32_bf16(false, a, false, b,
                                                (short)0, c, false, false);
  }
  return c;
}

// ---- 4 row-tiles (64 rows) GEMM with B-fragment register reuse ----
static __device__ __forceinline__ void wmma_rows64(
    const __bf16* ldsA, int strideA, int KT,
    const __bf16* Bp, int ct, float bv, v8f C[4]) {
  int lane = threadIdx.x & 31;
  int half = lane >> 4;
  int row  = lane & 15;
#pragma unroll
  for (int rt = 0; rt < 4; ++rt)
#pragma unroll
    for (int r = 0; r < 8; ++r) C[rt][r] = bv;
  const __bf16* arow = ldsA + row * strideA + half * 8;
  const __bf16* bp   = Bp + (size_t)ct * KT * 512 + lane * 16;
#pragma unroll 4
  for (int kt = 0; kt < KT; ++kt) {
    v16bf b = *(const v16bf*)(bp + kt * 512);   // load B once ...
#pragma unroll
    for (int rt = 0; rt < 4; ++rt) {            // ... reuse for 4 row tiles
      v16bf a = load_afrag(arow + rt * 16 * strideA + kt * 32);
      C[rt] = __builtin_amdgcn_wmma_f32_16x16x32_bf16(false, a, false, b,
                                                      (short)0, C[rt], false, false);
    }
  }
}

static __device__ __forceinline__ void store_tile_bf16_relu(
    __bf16* h, int stride, int rbase, int ct, const v8f& c) {
  int lane = threadIdx.x & 31;
  int half = lane >> 4, col = lane & 15;
#pragma unroll
  for (int r = 0; r < 8; ++r) {
    float v = c[r] > 0.f ? c[r] : 0.f;
    h[(rbase + r + half * 8) * stride + ct * 16 + col] = f2bf(v);
  }
}

static __device__ __forceinline__ void store_tile_f32(
    float* h, int stride, int rbase, int ct, const v8f& c) {
  int lane = threadIdx.x & 31;
  int half = lane >> 4, col = lane & 15;
#pragma unroll
  for (int r = 0; r < 8; ++r)
    h[(rbase + r + half * 8) * stride + ct * 16 + col] = c[r];
}

static __device__ __forceinline__ void store_tile_f32_relu(
    float* h, int stride, int rbase, int ct, const v8f& c) {
  int lane = threadIdx.x & 31;
  int half = lane >> 4, col = lane & 15;
#pragma unroll
  for (int r = 0; r < 8; ++r) {
    float v = c[r] > 0.f ? c[r] : 0.f;
    h[(rbase + r + half * 8) * stride + ct * 16 + col] = v;
  }
}

// LayerNorm stats for a [16][stride] f32 tile, 16 threads/row.
static __device__ __forceinline__ void ln_stats16(
    const float* h, int stride, float* red, float* mv, int tid) {
  int r = tid >> 4, s = tid & 15;
  float sum = 0.f, ss = 0.f;
#pragma unroll
  for (int j = 0; j < 8; ++j) {
    float v = h[r * stride + s * 8 + j];
    sum += v; ss += v * v;
  }
  red[r * 16 + s] = sum;
  red[256 + r * 16 + s] = ss;
  __syncthreads();
  if (s == 0) {
    float S = 0.f, S2 = 0.f;
#pragma unroll
    for (int q = 0; q < 16; ++q) { S += red[r * 16 + q]; S2 += red[256 + r * 16 + q]; }
    float mu  = S * (1.f / 128.f);
    float var = S2 * (1.f / 128.f) - mu * mu;
    mv[r * 2] = mu;
    mv[r * 2 + 1] = rsqrtf(var + 1e-5f);
  }
  __syncthreads();
}

// LayerNorm stats for a [64][stride] f32 tile, 4 threads/row (32 cols each).
static __device__ __forceinline__ void ln_stats64(
    const float* h, int stride, float* red, float* mv, int tid) {
  int r = tid >> 2, s = tid & 3;
  float sum = 0.f, ss = 0.f;
#pragma unroll
  for (int j = 0; j < 32; ++j) {
    float v = h[r * stride + s * 32 + j];
    sum += v; ss += v * v;
  }
  red[r * 4 + s] = sum;
  red[256 + r * 4 + s] = ss;
  __syncthreads();
  if (s == 0) {
    float S = 0.f, S2 = 0.f;
#pragma unroll
    for (int q = 0; q < 4; ++q) { S += red[r * 4 + q]; S2 += red[256 + r * 4 + q]; }
    float mu  = S * (1.f / 128.f);
    float var = S2 * (1.f / 128.f) - mu * mu;
    mv[r * 2] = mu;
    mv[r * 2 + 1] = rsqrtf(var + 1e-5f);
  }
  __syncthreads();
}

// ---------- weight prepack: f32 [K][128] -> fragment-major bf16 ----------
__global__ __launch_bounds__(256) void prepack_b(
    const float* __restrict__ src, __bf16* __restrict__ dst, int K) {
  int t = blockIdx.x * 256 + threadIdx.x;      // K*HH elements, exact multiple of 256
  int j    = t & 15;
  int lane = (t >> 4) & 31;
  int frag = t >> 9;                            // = ct*(K/32) + kt
  int KT = K >> 5;
  int kt = frag % KT;
  int ct = frag / KT;
  int k   = kt * 32 + ((j >> 3) << 4) + ((lane >> 4) << 3) + (j & 7);   // B layout: k = 16*(j>>3) + 8*half + (j&7)
  int col = ct * 16 + (lane & 15);
  dst[t] = f2bf(src[(size_t)k * HH + col]);
}

__global__ __launch_bounds__(256) void zero_kernel(float* __restrict__ p, int n) {
  int i = (blockIdx.x * 256 + threadIdx.x) * 4;
  if (i < n) { p[i] = 0.f; p[i + 1] = 0.f; p[i + 2] = 0.f; p[i + 3] = 0.f; }
}

// ---------- encoders: normalize -> (K=din scalar GEMM,ReLU) -> WMMA -> LN ----------
__global__ __launch_bounds__(256) void encoder_kernel(
    const float* __restrict__ in, int din,
    const float* __restrict__ mean, const float* __restrict__ stdv,
    const float* __restrict__ w1, const float* __restrict__ b1,
    const __bf16* __restrict__ w2p, const float* __restrict__ b2,
    const float* __restrict__ g, const float* __restrict__ bb,
    float* __restrict__ out) {
  __shared__ float  xin[16][12];
  __shared__ __bf16 h1[16][136];
  __shared__ float  h2[16][132];
  __shared__ float  red[512];
  __shared__ float  mv[32];
  int tid = threadIdx.x;
  int base = blockIdx.x * 16;
  for (int idx = tid; idx < 16 * din; idx += 256) {
    int r = idx / din, c0 = idx % din;
    xin[r][c0] = (in[(size_t)(base + r) * din + c0] - mean[c0]) / stdv[c0];
  }
  __syncthreads();
  {
    int r = tid >> 4, cb = (tid & 15) * 8;
#pragma unroll
    for (int j = 0; j < 8; ++j) {
      int c = cb + j;
      float acc = b1[c];
      for (int k = 0; k < din; ++k) acc += xin[r][k] * w1[k * HH + c];
      h1[r][c] = f2bf(acc > 0.f ? acc : 0.f);
    }
  }
  __syncthreads();
  {
    int ct = tid >> 5;
    float bv = b2[ct * 16 + (tid & 15)];
    v8f c = wmma_rows16(&h1[0][0], 136, 4, w2p, ct, bv);
    store_tile_f32(&h2[0][0], 132, 0, ct, c);
  }
  __syncthreads();
  ln_stats16(&h2[0][0], 132, red, mv, tid);
  {
    int r = tid >> 4, cb = (tid & 15) * 8;
    float mu = mv[r * 2], rs = mv[r * 2 + 1];
#pragma unroll
    for (int j = 0; j < 8; ++j) {
      int c = cb + j;
      out[(size_t)(base + r) * HH + c] = (h2[r][c] - mu) * rs * g[c] + bb[c];
    }
  }
}

// ---------- edge message: MLP_LN(cat[x_dst,x_src,ea]) + ea ; atomic agg ----------
// 64 edges per block; 8 waves x (1 col tile, 4 row tiles).
__global__ __launch_bounds__(256) void edge_kernel(
    const float* __restrict__ x, const float* __restrict__ ea_in,
    const int* __restrict__ ei,
    const __bf16* __restrict__ w1p, const float* __restrict__ b1,
    const __bf16* __restrict__ w2p, const float* __restrict__ b2,
    const float* __restrict__ g, const float* __restrict__ bb,
    float* __restrict__ ea_out, float* __restrict__ agg) {
  __shared__ __bf16 A[64][392];     // row stride 784B -> 4-bank skew, conflict free
  __shared__ __bf16 h1[64][136];
  __shared__ float  h2[64][132];
  __shared__ float  red[512];
  __shared__ float  mv[128];
  int tid = threadIdx.x;
  int e0 = blockIdx.x * 64;
  int r = tid >> 2, s = tid & 3;    // 4 threads per row, 32 cols each
  {
    int e = e0 + r;
    int srcn = ei[e];          // edge_index[0] = j (also aggregation index)
    int dstn = ei[EE + e];     // edge_index[1] = i
    const float* xd  = x + (size_t)dstn * HH;
    const float* xs  = x + (size_t)srcn * HH;
    const float* ear = ea_in + (size_t)e * HH;
#pragma unroll
    for (int j = 0; j < 32; ++j) {
      int c = s * 32 + j;
      A[r][c]          = f2bf(xd[c]);
      A[r][HH + c]     = f2bf(xs[c]);
      A[r][2 * HH + c] = f2bf(ear[c]);
    }
  }
  __syncthreads();
  {
    int ct = tid >> 5;
    float bv = b1[ct * 16 + (tid & 15)];
    v8f C[4];
    wmma_rows64(&A[0][0], 392, 12, w1p, ct, bv, C);
#pragma unroll
    for (int rt = 0; rt < 4; ++rt)
      store_tile_bf16_relu(&h1[0][0], 136, rt * 16, ct, C[rt]);
  }
  __syncthreads();
  {
    int ct = tid >> 5;
    float bv = b2[ct * 16 + (tid & 15)];
    v8f C[4];
    wmma_rows64(&h1[0][0], 136, 4, w2p, ct, bv, C);
#pragma unroll
    for (int rt = 0; rt < 4; ++rt)
      store_tile_f32(&h2[0][0], 132, rt * 16, ct, C[rt]);
  }
  __syncthreads();
  ln_stats64(&h2[0][0], 132, red, mv, tid);
  {
    int e = e0 + r;
    int srcn = ei[e];
    float mu = mv[r * 2], rs = mv[r * 2 + 1];
    float* aggr = agg + (size_t)srcn * HH;
#pragma unroll
    for (int j = 0; j < 32; ++j) {
      int c = s * 32 + j;
      float m = (h2[r][c] - mu) * rs * g[c] + bb[c] + ea_in[(size_t)e * HH + c];
      ea_out[(size_t)e * HH + c] = m;
      unsafeAtomicAdd(&aggr[c], m);   // global_atomic_add_f32, agg is L2-resident
    }
  }
}

// ---------- node update: x += MLP_LN(cat[x, agg]) ; 64 nodes per block ----------
__global__ __launch_bounds__(256) void node_kernel(
    float* __restrict__ x, const float* __restrict__ agg,
    const __bf16* __restrict__ w1p, const float* __restrict__ b1,
    const __bf16* __restrict__ w2p, const float* __restrict__ b2,
    const float* __restrict__ g, const float* __restrict__ bb) {
  __shared__ __bf16 A[64][264];
  __shared__ __bf16 h1[64][136];
  __shared__ float  h2[64][132];
  __shared__ float  red[512];
  __shared__ float  mv[128];
  int tid = threadIdx.x;
  int n0 = blockIdx.x * 64;
  int r = tid >> 2, s = tid & 3;
  {
    const float* xr = x + (size_t)(n0 + r) * HH;
    const float* ar = agg + (size_t)(n0 + r) * HH;
#pragma unroll
    for (int j = 0; j < 32; ++j) {
      int c = s * 32 + j;
      A[r][c]      = f2bf(xr[c]);
      A[r][HH + c] = f2bf(ar[c]);
    }
  }
  __syncthreads();
  {
    int ct = tid >> 5;
    float bv = b1[ct * 16 + (tid & 15)];
    v8f C[4];
    wmma_rows64(&A[0][0], 264, 8, w1p, ct, bv, C);
#pragma unroll
    for (int rt = 0; rt < 4; ++rt)
      store_tile_bf16_relu(&h1[0][0], 136, rt * 16, ct, C[rt]);
  }
  __syncthreads();
  {
    int ct = tid >> 5;
    float bv = b2[ct * 16 + (tid & 15)];
    v8f C[4];
    wmma_rows64(&h1[0][0], 136, 4, w2p, ct, bv, C);
#pragma unroll
    for (int rt = 0; rt < 4; ++rt)
      store_tile_f32(&h2[0][0], 132, rt * 16, ct, C[rt]);
  }
  __syncthreads();
  ln_stats64(&h2[0][0], 132, red, mv, tid);
  {
    float mu = mv[r * 2], rs = mv[r * 2 + 1];
    float* xr = x + (size_t)(n0 + r) * HH;
#pragma unroll
    for (int j = 0; j < 32; ++j) {
      int c = s * 32 + j;
      xr[c] = xr[c] + (h2[r][c] - mu) * rs * g[c] + bb[c];
    }
  }
}

// ---------- decoder: relu(x@w1+b1)@w2+b2 ----------
__global__ __launch_bounds__(256) void decoder_kernel(
    const float* __restrict__ x,
    const __bf16* __restrict__ w1p, const float* __restrict__ b1,
    const float* __restrict__ w2, const float* __restrict__ b2,
    float* __restrict__ out) {
  __shared__ __bf16 A[16][136];
  __shared__ float  h1[16][132];
  int tid = threadIdx.x;
  int n0 = blockIdx.x * 16;
  int r = tid >> 4, s = tid & 15;
  {
    const float* xr = x + (size_t)(n0 + r) * HH;
#pragma unroll
    for (int j = 0; j < 8; ++j) {
      int c = s * 8 + j;
      A[r][c] = f2bf(xr[c]);
    }
  }
  __syncthreads();
  {
    int ct = tid >> 5;
    float bv = b1[ct * 16 + (tid & 15)];
    v8f c = wmma_rows16(&A[0][0], 136, 4, w1p, ct, bv);
    store_tile_f32_relu(&h1[0][0], 132, 0, ct, c);
  }
  __syncthreads();
  if (tid < 16 * DOUT) {
    int rr = tid / DOUT, d = tid % DOUT;
    float acc = b2[d];
    for (int k = 0; k < HH; ++k) acc += h1[rr][k] * w2[k * DOUT + d];
    out[(size_t)(n0 + rr) * DOUT + d] = acc;
  }
}

// ---------- host ----------
extern "C" void kernel_launch(void* const* d_in, const int* in_sizes, int n_in,
                              void* d_out, int out_size, void* d_ws, size_t ws_size,
                              hipStream_t stream) {
  (void)in_sizes; (void)n_in; (void)out_size; (void)ws_size;
  const float* x_in   = (const float*)d_in[0];
  const float* eattr  = (const float*)d_in[1];
  const int*   ei     = (const int*)d_in[2];
  const float* mean_x = (const float*)d_in[3];
  const float* std_x  = (const float*)d_in[4];
  const float* mean_e = (const float*)d_in[5];
  const float* std_e  = (const float*)d_in[6];
  const float* en_w1  = (const float*)d_in[7];
  const float* en_b1  = (const float*)d_in[8];
  const float* en_w2  = (const float*)d_in[9];
  const float* en_b2  = (const float*)d_in[10];
  const float* en_g   = (const float*)d_in[11];
  const float* en_b   = (const float*)d_in[12];
  const float* ee_w1  = (const float*)d_in[13];
  const float* ee_b1  = (const float*)d_in[14];
  const float* ee_w2  = (const float*)d_in[15];
  const float* ee_b2  = (const float*)d_in[16];
  const float* ee_g   = (const float*)d_in[17];
  const float* ee_b   = (const float*)d_in[18];
  const float* pn_w1  = (const float*)d_in[19];
  const float* pn_b1  = (const float*)d_in[20];
  const float* pn_w2  = (const float*)d_in[21];
  const float* pn_b2  = (const float*)d_in[22];
  const float* pn_g   = (const float*)d_in[23];
  const float* pn_b   = (const float*)d_in[24];
  const float* pe_w1  = (const float*)d_in[25];
  const float* pe_b1  = (const float*)d_in[26];
  const float* pe_w2  = (const float*)d_in[27];
  const float* pe_b2  = (const float*)d_in[28];
  const float* pe_g   = (const float*)d_in[29];
  const float* pe_b   = (const float*)d_in[30];
  const float* dec_w1 = (const float*)d_in[31];
  const float* dec_b1 = (const float*)d_in[32];
  const float* dec_w2 = (const float*)d_in[33];
  const float* dec_b2 = (const float*)d_in[34];

  char* wp = (char*)d_ws;
  auto take = [&](size_t bytes) {
    char* p = wp;
    wp += (bytes + 255) & ~(size_t)255;
    return p;
  };
  float*  X      = (float*)take((size_t)NN * HH * 4);
  float*  EA0    = (float*)take((size_t)EE * HH * 4);
  float*  EA1    = (float*)take((size_t)EE * HH * 4);
  float*  AGG    = (float*)take((size_t)NN * HH * 4);
  __bf16* p_enw2 = (__bf16*)take((size_t)HH * HH * 2);
  __bf16* p_eew2 = (__bf16*)take((size_t)HH * HH * 2);
  __bf16* p_dw1  = (__bf16*)take((size_t)HH * HH * 2);
  __bf16* p_pew1 = (__bf16*)take((size_t)LL * 3 * HH * HH * 2);
  __bf16* p_pew2 = (__bf16*)take((size_t)LL * HH * HH * 2);
  __bf16* p_pnw1 = (__bf16*)take((size_t)LL * 2 * HH * HH * 2);
  __bf16* p_pnw2 = (__bf16*)take((size_t)LL * HH * HH * 2);

  auto pack = [&](const float* src, __bf16* dst, int K) {
    prepack_b<<<(K * HH + 255) / 256, 256, 0, stream>>>(src, dst, K);
  };
  pack(en_w2, p_enw2, 128);
  pack(ee_w2, p_eew2, 128);
  pack(dec_w1, p_dw1, 128);
  for (int l = 0; l < LL; ++l) {
    pack(pe_w1 + (size_t)l * 3 * HH * HH, p_pew1 + (size_t)l * 3 * HH * HH, 384);
    pack(pe_w2 + (size_t)l * HH * HH,     p_pew2 + (size_t)l * HH * HH,     128);
    pack(pn_w1 + (size_t)l * 2 * HH * HH, p_pnw1 + (size_t)l * 2 * HH * HH, 256);
    pack(pn_w2 + (size_t)l * HH * HH,     p_pnw2 + (size_t)l * HH * HH,     128);
  }

  encoder_kernel<<<NN / 16, 256, 0, stream>>>(x_in, DIN, mean_x, std_x,
      en_w1, en_b1, p_enw2, en_b2, en_g, en_b, X);
  encoder_kernel<<<EE / 16, 256, 0, stream>>>(eattr, DEDGE, mean_e, std_e,
      ee_w1, ee_b1, p_eew2, ee_b2, ee_g, ee_b, EA0);

  float* ea_in = EA0;
  float* ea_out = EA1;
  for (int l = 0; l < LL; ++l) {
    zero_kernel<<<(NN * HH / 4 + 255) / 256, 256, 0, stream>>>(AGG, NN * HH);
    edge_kernel<<<EE / 64, 256, 0, stream>>>(X, ea_in, ei,
        p_pew1 + (size_t)l * 3 * HH * HH, pe_b1 + l * HH,
        p_pew2 + (size_t)l * HH * HH,     pe_b2 + l * HH,
        pe_g + l * HH, pe_b + l * HH, ea_out, AGG);
    node_kernel<<<NN / 64, 256, 0, stream>>>(X, AGG,
        p_pnw1 + (size_t)l * 2 * HH * HH, pn_b1 + l * HH,
        p_pnw2 + (size_t)l * HH * HH,     pn_b2 + l * HH,
        pn_g + l * HH, pn_b + l * HH);
    float* t = ea_in; ea_in = ea_out; ea_out = t;
  }

  decoder_kernel<<<NN / 16, 256, 0, stream>>>(X, p_dw1, dec_b1,
      dec_w2, dec_b2, (float*)d_out);
}
